// SimpleSelfAttention_8821862826339
// MI455X (gfx1250) — matrix operand order repaired and checked by reading, not compiled, not run on hardware
//
#include <hip/hip_runtime.h>
#include <hip/hip_bf16.h>

typedef __attribute__((ext_vector_type(16))) _Float16 v16h;
typedef __attribute__((ext_vector_type(8)))  _Float16 v8h;
typedef __attribute__((ext_vector_type(4)))  _Float16 v4h;
typedef __attribute__((ext_vector_type(8)))  float    v8f;

#define EMBED    1024
#define HEADS    16
#define HEAD_DIM 64
#define SEQ      2048
#define BATCH    2
#define ROWS     (BATCH * SEQ)      // 4096 token rows

static __device__ __forceinline__ v16h combine(v8h lo, v8h hi) {
    v16h r;
#pragma unroll
    for (int i = 0; i < 8; ++i) { r[i] = lo[i]; r[i + 8] = hi[i]; }
    return r;
}

static __device__ __forceinline__ v16h ld2(const _Float16* p0, const _Float16* p1) {
    return combine(*(const v8h*)p0, *(const v8h*)p1);
}

static __device__ __forceinline__ v8f wmma_f16(v16h a, v16h b, v8f c) {
    // D(f32 16x16) = A(f16 16x32) * B(f16 32x16) + C
    return __builtin_amdgcn_wmma_f32_16x16x32_f16(false, a, false, b, (short)0, c, false, false);
}

static __device__ __forceinline__ v8f zero8() {
    v8f z;
#pragma unroll
    for (int i = 0; i < 8; ++i) z[i] = 0.0f;
    return z;
}

// ---------------------------------------------------------------------------
// Kernel 1: QKV projection.  qkv = x @ w_qkv + b_qkv  (M=4096, K=1024, N=3072)
// Block: 128 threads (4 waves); tile 64x64; K staged through LDS in f16.
// Q (pre-scaled by 1/sqrt(d)), K stored [n][h][s][d] f16; V stored [n][h][d][s].
// ---------------------------------------------------------------------------
__global__ __launch_bounds__(128)
void qkv_gemm_kernel(const float* __restrict__ x, const float* __restrict__ w,
                     const float* __restrict__ bias,
                     _Float16* __restrict__ q_ws, _Float16* __restrict__ k_ws,
                     _Float16* __restrict__ v_ws) {
    const int j0    = blockIdx.x * 64;   // output column base (0..3071)
    const int Mbase = blockIdx.y * 64;   // token-row base (0..4095)
    const int tid   = threadIdx.x;
    const int wid   = tid >> 5;
    const int lane  = tid & 31;
    const int hi    = lane >> 4;
    const int cl    = lane & 15;
    const int W     = 3 * EMBED;

    __shared__ alignas(16) _Float16 Ab[64][32];   // rows x k-chunk
    __shared__ alignas(16) _Float16 Bb[64][32];   // cols x k-chunk (transposed stage)

    v8f acc[4];
#pragma unroll
    for (int c = 0; c < 4; ++c) acc[c] = zero8();

    for (int k0 = 0; k0 < EMBED; k0 += 32) {
        // ---- batched vector loads (all issued before any use) ----
        float4 aL[4], bL[4];
#pragma unroll
        for (int i = 0; i < 4; ++i) {            // A: 64x32 fp32, 16B per thread x4
            const int idx4 = tid + 128 * i;      // 0..511 float4 chunks
            const int r = idx4 >> 3, c = (idx4 & 7) * 4;
            aL[i] = *(const float4*)&x[(size_t)(Mbase + r) * EMBED + k0 + c];
        }
#pragma unroll
        for (int i = 0; i < 4; ++i) {            // B: 32x64 fp32, coalesced on cols
            const int idx4 = tid + 128 * i;
            const int k = idx4 >> 4, c4 = (idx4 & 15) * 4;
            bL[i] = *(const float4*)&w[(size_t)(k0 + k) * W + j0 + c4];
        }
        // ---- convert + store to LDS ----
#pragma unroll
        for (int i = 0; i < 4; ++i) {
            const int idx4 = tid + 128 * i;
            const int r = idx4 >> 3, c = (idx4 & 7) * 4;
            v4h hv;
            hv[0] = (_Float16)aL[i].x; hv[1] = (_Float16)aL[i].y;
            hv[2] = (_Float16)aL[i].z; hv[3] = (_Float16)aL[i].w;
            *(v4h*)&Ab[r][c] = hv;               // ds_store_b64
        }
#pragma unroll
        for (int i = 0; i < 4; ++i) {
            const int idx4 = tid + 128 * i;
            const int k = idx4 >> 4, c4 = (idx4 & 15) * 4;
            Bb[c4 + 0][k] = (_Float16)bL[i].x;
            Bb[c4 + 1][k] = (_Float16)bL[i].y;
            Bb[c4 + 2][k] = (_Float16)bL[i].z;
            Bb[c4 + 3][k] = (_Float16)bL[i].w;
        }
        __syncthreads();

        const int arow = 16 * wid + cl;
        v16h a = ld2(&Ab[arow][16 * hi], &Ab[arow][16 * hi + 8]);
#pragma unroll
        for (int c = 0; c < 4; ++c) {
            const int col = 16 * c + cl;
            v16h b = ld2(&Bb[col][16 * hi], &Bb[col][16 * hi + 8]);
            acc[c] = wmma_f16(a, b, acc[c]);
        }
        __syncthreads();
    }

    // ---- bias + scatter.  part is block-uniform (64-col tile within one of Q/K/V)
    const int part = j0 >> 10;                   // 0=Q 1=K 2=V
    const float qscale = (part == 0) ? 0.125f : 1.0f;  // fold 1/sqrt(64) into Q
#pragma unroll
    for (int c = 0; c < 4; ++c) {
        const int j    = j0 + 16 * c + cl;
        const float bj = bias[j];
        const int wi   = j & (EMBED - 1);
        const int hh   = wi >> 6;
        const int dd   = wi & (HEAD_DIM - 1);
#pragma unroll
        for (int v = 0; v < 8; ++v) {
            const int M  = Mbase + 16 * wid + v + 8 * hi;
            const int nb = M >> 11;
            const int ss = M & (SEQ - 1);
            const _Float16 hv = (_Float16)((acc[c][v] + bj) * qscale);
            if (part == 0) {
                q_ws[((size_t)(nb * HEADS + hh) * SEQ + ss) * HEAD_DIM + dd] = hv;
            } else if (part == 1) {
                k_ws[((size_t)(nb * HEADS + hh) * SEQ + ss) * HEAD_DIM + dd] = hv;
            } else {
                v_ws[((size_t)(nb * HEADS + hh) * HEAD_DIM + dd) * SEQ + ss] = hv;
            }
        }
    }
}

// ---------------------------------------------------------------------------
// Kernel 2: flash attention, one wave per (batch, head, 16-query block).
// Online softmax over key blocks of 32; QK^T and P*V both via WMMA f16.
// Q already carries the 1/sqrt(d) scale.
// ---------------------------------------------------------------------------
__global__ __launch_bounds__(32)
void attn_kernel(const _Float16* __restrict__ q_ws, const _Float16* __restrict__ k_ws,
                 const _Float16* __restrict__ v_ws, _Float16* __restrict__ attnout) {
    const int qblk = blockIdx.x;        // 0..127
    const int h    = blockIdx.y;        // 0..15
    const int nb   = blockIdx.z;        // 0..1
    const int lane = threadIdx.x;
    const int hi   = lane >> 4;
    const int cl   = lane & 15;

    const _Float16* qbase = q_ws + ((size_t)(nb * HEADS + h) * SEQ + qblk * 16) * HEAD_DIM;
    const _Float16* kbase = k_ws + (size_t)(nb * HEADS + h) * SEQ * HEAD_DIM;
    const _Float16* vbase = v_ws + (size_t)(nb * HEADS + h) * HEAD_DIM * SEQ;

    // Q tile (16x64) held in registers as two 16x32 A-operands
    const _Float16* qr = qbase + (size_t)cl * HEAD_DIM;
    const v16h aq0 = ld2(qr + 16 * hi,      qr + 16 * hi + 8);
    const v16h aq1 = ld2(qr + 32 + 16 * hi, qr + 32 + 16 * hi + 8);

    float m[8], l[8];
    v8f o[4];
#pragma unroll
    for (int v = 0; v < 8; ++v) { m[v] = -1e30f; l[v] = 0.0f; }
#pragma unroll
    for (int t = 0; t < 4; ++t) o[t] = zero8();

    __shared__ alignas(16) _Float16 Pb[16][32];

    for (int key0 = 0; key0 < SEQ; key0 += 32) {
        // ---- scores: two 16x16 tiles (keys key0..+15, key0+16..+31) ----
        const _Float16* kt0 = kbase + (size_t)(key0 + cl) * HEAD_DIM;
        const _Float16* kt1 = kbase + (size_t)(key0 + 16 + cl) * HEAD_DIM;
        v16h bk00 = ld2(kt0 + 16 * hi,      kt0 + 16 * hi + 8);       // d 0..31
        v16h bk01 = ld2(kt0 + 32 + 16 * hi, kt0 + 32 + 16 * hi + 8);  // d 32..63
        v16h bk10 = ld2(kt1 + 16 * hi,      kt1 + 16 * hi + 8);
        v16h bk11 = ld2(kt1 + 32 + 16 * hi, kt1 + 32 + 16 * hi + 8);

        v8f ct0 = wmma_f16(aq0, bk00, zero8());
        ct0     = wmma_f16(aq1, bk01, ct0);
        v8f ct1 = wmma_f16(aq0, bk10, zero8());
        ct1     = wmma_f16(aq1, bk11, ct1);

        float corr[8];
#pragma unroll
        for (int v = 0; v < 8; ++v) {
            const float s0 = ct0[v];
            const float s1 = ct1[v];
            // row max across the 16 lanes holding this row
            float rm = fmaxf(s0, s1);
            rm = fmaxf(rm, __shfl_xor(rm, 1, 32));
            rm = fmaxf(rm, __shfl_xor(rm, 2, 32));
            rm = fmaxf(rm, __shfl_xor(rm, 4, 32));
            rm = fmaxf(rm, __shfl_xor(rm, 8, 32));
            const float mn = fmaxf(m[v], rm);
            corr[v] = __expf(m[v] - mn);
            m[v] = mn;
            const float p0 = __expf(s0 - mn);
            const float p1 = __expf(s1 - mn);
            float rs = p0 + p1;
            rs += __shfl_xor(rs, 1, 32);
            rs += __shfl_xor(rs, 2, 32);
            rs += __shfl_xor(rs, 4, 32);
            rs += __shfl_xor(rs, 8, 32);
            l[v] = l[v] * corr[v] + rs;
            Pb[v + 8 * hi][cl]      = (_Float16)p0;
            Pb[v + 8 * hi][cl + 16] = (_Float16)p1;
        }
#pragma unroll
        for (int t = 0; t < 4; ++t)
#pragma unroll
            for (int v = 0; v < 8; ++v) o[t][v] *= corr[v];

        __syncthreads();   // order P store -> A-pattern reload
        v16h pa = ld2(&Pb[cl][16 * hi], &Pb[cl][16 * hi + 8]);

        // ---- P (16x32) * V (32keys x 64d) via 4 WMMA over d tiles ----
#pragma unroll
        for (int t = 0; t < 4; ++t) {
            const _Float16* vp = vbase + (size_t)(16 * t + cl) * SEQ + key0 + 16 * hi;
            v16h bv = ld2(vp, vp + 8);
            o[t] = wmma_f16(pa, bv, o[t]);
        }
        __syncthreads();   // protect Pb before next iteration overwrites it
    }

    // normalize + store f16 attention output [n][s][h*64+d]
#pragma unroll
    for (int v = 0; v < 8; ++v) {
        const float inv = 1.0f / l[v];
        const int q = qblk * 16 + v + 8 * hi;
#pragma unroll
        for (int t = 0; t < 4; ++t) {
            attnout[(size_t)(nb * SEQ + q) * EMBED + h * HEAD_DIM + 16 * t + cl] =
                (_Float16)(o[t][v] * inv);
        }
    }
}

// ---------------------------------------------------------------------------
// Kernel 3: output projection.  out = attn @ w_out + b_out  (M=4096, K=N=1024)
// ---------------------------------------------------------------------------
__global__ __launch_bounds__(128)
void out_gemm_kernel(const _Float16* __restrict__ a_in, const float* __restrict__ w,
                     const float* __restrict__ bias, float* __restrict__ out) {
    const int j0    = blockIdx.x * 64;
    const int Mbase = blockIdx.y * 64;
    const int tid   = threadIdx.x;
    const int wid   = tid >> 5;
    const int lane  = tid & 31;
    const int hi    = lane >> 4;
    const int cl    = lane & 15;

    __shared__ alignas(16) _Float16 Ab[64][32];
    __shared__ alignas(16) _Float16 Bb[64][32];

    v8f acc[4];
#pragma unroll
    for (int c = 0; c < 4; ++c) acc[c] = zero8();

    for (int k0 = 0; k0 < EMBED; k0 += 32) {
        // ---- batched vector loads ----
        v8h  aL[2];
        float4 bL[4];
#pragma unroll
        for (int i = 0; i < 2; ++i) {            // A: 64x32 f16 copy, 16B x2 per thread
            const int idx8 = tid + 128 * i;      // 0..255 v8h chunks
            const int r = idx8 >> 2, c = (idx8 & 3) * 8;
            aL[i] = *(const v8h*)&a_in[(size_t)(Mbase + r) * EMBED + k0 + c];
        }
#pragma unroll
        for (int i = 0; i < 4; ++i) {            // B: 32x64 fp32, coalesced on cols
            const int idx4 = tid + 128 * i;
            const int k = idx4 >> 4, c4 = (idx4 & 15) * 4;
            bL[i] = *(const float4*)&w[(size_t)(k0 + k) * EMBED + j0 + c4];
        }
#pragma unroll
        for (int i = 0; i < 2; ++i) {
            const int idx8 = tid + 128 * i;
            const int r = idx8 >> 2, c = (idx8 & 3) * 8;
            *(v8h*)&Ab[r][c] = aL[i];            // ds_store_b128
        }
#pragma unroll
        for (int i = 0; i < 4; ++i) {
            const int idx4 = tid + 128 * i;
            const int k = idx4 >> 4, c4 = (idx4 & 15) * 4;
            Bb[c4 + 0][k] = (_Float16)bL[i].x;
            Bb[c4 + 1][k] = (_Float16)bL[i].y;
            Bb[c4 + 2][k] = (_Float16)bL[i].z;
            Bb[c4 + 3][k] = (_Float16)bL[i].w;
        }
        __syncthreads();

        const int arow = 16 * wid + cl;
        v16h a = ld2(&Ab[arow][16 * hi], &Ab[arow][16 * hi + 8]);
#pragma unroll
        for (int c = 0; c < 4; ++c) {
            const int col = 16 * c + cl;
            v16h b = ld2(&Bb[col][16 * hi], &Bb[col][16 * hi + 8]);
            acc[c] = wmma_f16(a, b, acc[c]);
        }
        __syncthreads();
    }

#pragma unroll
    for (int c = 0; c < 4; ++c) {
        const int j = j0 + 16 * c + cl;
        const float bj = bias[j];
#pragma unroll
        for (int v = 0; v < 8; ++v) {
            const int M = Mbase + 16 * wid + v + 8 * hi;
            out[(size_t)M * EMBED + j] = acc[c][v] + bj;
        }
    }
}

// ---------------------------------------------------------------------------
extern "C" void kernel_launch(void* const* d_in, const int* in_sizes, int n_in,
                              void* d_out, int out_size, void* d_ws, size_t ws_size,
                              hipStream_t stream) {
    const float* x     = (const float*)d_in[0];
    const float* w_qkv = (const float*)d_in[1];
    const float* b_qkv = (const float*)d_in[2];
    const float* w_out = (const float*)d_in[3];
    const float* b_out = (const float*)d_in[4];
    float* out = (float*)d_out;

    const size_t HSD = (size_t)BATCH * HEADS * SEQ * HEAD_DIM;  // 4,194,304 elems
    _Float16* q_ws    = (_Float16*)d_ws;
    _Float16* k_ws    = q_ws + HSD;
    _Float16* v_ws    = k_ws + HSD;     // transposed [n][h][d][s]
    _Float16* attnout = v_ws + HSD;     // [n][s][e] f16

    // 1) QKV projection -> f16 Q(pre-scaled), K, V^T
    {
        dim3 grid(3 * EMBED / 64, ROWS / 64);   // 48 x 64
        qkv_gemm_kernel<<<grid, 128, 0, stream>>>(x, w_qkv, b_qkv, q_ws, k_ws, v_ws);
    }
    // 2) attention (flash online softmax, one wave per 16-query block)
    {
        dim3 grid(SEQ / 16, HEADS, BATCH);      // 128 x 16 x 2
        attn_kernel<<<grid, 32, 0, stream>>>(q_ws, k_ws, v_ws, attnout);
    }
    // 3) output projection -> fp32
    {
        dim3 grid(EMBED / 64, ROWS / 64);       // 16 x 64
        out_gemm_kernel<<<grid, 128, 0, stream>>>(attnout, w_out, b_out, out);
    }
}
